// Mamba_align_11690900980490
// MI455X (gfx1250) — matrix operand order repaired and checked by reading, not compiled, not run on hardware
//
#include <hip/hip_runtime.h>

// ---------------------------------------------------------------------------
// MI455X (gfx1250): WMMA f16->f32 for all GEMM/conv math, chunked parallel
// selective scan, global_prefetch on the scan streams.
// ---------------------------------------------------------------------------

typedef _Float16 f16;
typedef __attribute__((ext_vector_type(16))) _Float16 v16h;
typedef __attribute__((ext_vector_type(8)))  float    v8f;

#define NEG_SLOPE 0.1f
#define F7   7
#define BB   2
#define DM   64
#define HH   64
#define WW   64
#define HW   4096          // 64*64
#define LL   28672         // 7*4096
#define BL   57344         // 2*28672
#define NIMG 14            // 7*2
#define CH   256           // scan chunk length
#define NC   112           // LL / CH

// workspace offsets (in floats)
#define OFF_FRAMES 0u            // 3670016  (7,2,64,64,64)
#define OFF_R      3670016u      // 3670016  conv3 output
#define OFF_XFLAT  7340032u      // 3670016  (BL,64)  -- later reused as ACT448
#define OFF_XZ     11010048u     // 7340032  (BL,128) -- later reused as M (BL,64)
#define OFF_XS     18350080u     // 3670016  (BL,64)  -- later reused as Y2
#define OFF_DT     22020096u     // 3670016  (BL,64)  -- later reused as YSCAN
#define OFF_BC     25690112u     // 1835008  (BL,32)  -- later reused as OUTT (8192,64)
#define OFF_WEFF   27525120u     // 4096     (64,64)
#define OFF_WT3    27529216u     // 18432 floats = 36864 f16  [tap][oc][ci=64]
#define OFF_WT13   27547648u     // 36864 floats = 73728 f16  [tap][oc][ci=128]
#define OFF_CP     27584512u     // 229376   per-chunk prod(dA)   (b,d,chunk,n)
#define OFF_CS     27813888u     // 229376   per-chunk partial S
#define OFF_HS     28043264u     // 229376   per-chunk entry state
// total = 28272640 floats = ~113 MB

__device__ __forceinline__ v8f wmma16(v16h a, v16h b, v8f c) {
  return __builtin_amdgcn_wmma_f32_16x16x32_f16(false, a, false, b, (short)0, c,
                                                false, false);
}

// A fragment (16x32, f16) from row-major fp32 A. lane: m = lane&15, half = lane>>4.
__device__ __forceinline__ v16h load_a_f32(const float* A, int lda, int mrow,
                                           int kbase, int lane) {
  int m = lane & 15, half = lane >> 4;
  v16h a;
#pragma unroll
  for (int v = 0; v < 8; ++v) {
    int k = kbase + half * 16 + 2 * v;
    const float* p = A + (size_t)(mrow + m) * lda + k;
    a[2 * v]     = (f16)p[0];
    a[2 * v + 1] = (f16)p[1];
  }
  return a;
}

// B fragment (32x16 = K x N) from weights W row-major (N,K): Bmat[k][n] = W[n][k].
__device__ __forceinline__ v16h load_b_w(const float* W, int ldw, int nrow,
                                         int kbase, int lane) {
  int n = lane & 15, half = lane >> 4;
  v16h b;
#pragma unroll
  for (int v = 0; v < 8; ++v) {
    int k = kbase + half * 16 + 2 * v;
    const float* p = W + (size_t)(nrow + n) * ldw + k;
    b[2 * v]     = (f16)p[0];
    b[2 * v + 1] = (f16)p[1];
  }
  return b;
}

// ---------------------------------------------------------------------------
// GEMM: C[m,0..NT*16) = epi( A[m,:] @ W^T + bias ). One wave per 16-row strip,
// A fragment loaded once per k-step and reused across NT column tiles.
// EPI: 0 none, 1 softplus(bias), 2 lrelu(bias).
// ---------------------------------------------------------------------------
template <int EPI, int NT>
__global__ void gemm_wmma(const float* __restrict__ A, const float* __restrict__ W,
                          const float* __restrict__ bias, float* __restrict__ C,
                          int K, int lda, int ldw, int ldc) {
  int tm = blockIdx.x;
  int lane = threadIdx.x & 31;
  v8f acc[NT];
#pragma unroll
  for (int t = 0; t < NT; ++t) acc[t] = (v8f){};
  for (int kb = 0; kb < K; kb += 32) {
    v16h a = load_a_f32(A, lda, tm * 16, kb, lane);
#pragma unroll
    for (int t = 0; t < NT; ++t) {
      v16h b = load_b_w(W, ldw, t * 16, kb, lane);
      acc[t] = wmma16(a, b, acc[t]);
    }
  }
  int n = lane & 15, half = lane >> 4;
#pragma unroll
  for (int t = 0; t < NT; ++t) {
    int col = t * 16 + n;
    float bv = (EPI != 0) ? bias[col] : 0.f;
#pragma unroll
    for (int v = 0; v < 8; ++v) {
      int m = tm * 16 + v + half * 8;
      float x = acc[t][v] + bv;
      if (EPI == 1) x = (x > 20.f) ? x : logf(1.f + expf(x));   // softplus
      if (EPI == 2) x = (x >= 0.f) ? x : NEG_SLOPE * x;          // lrelu
      C[(size_t)m * ldc + col] = x;
    }
  }
}

// ---------------------------------------------------------------------------
// 3x3 conv implicit GEMM. Block = 128 threads (4 waves), each wave 16 out-ch.
// LDS patch: [3][18][CI] f16. MODE 0: conv3 (lrelu). MODE 1: conv13
// (channels = [r | outT], lrelu + residual frames, write d_out).
// ---------------------------------------------------------------------------
template <int CI, int MODE>
__global__ void conv3x3_wmma(const float* __restrict__ src0,   // r or frames (n,c,h,w)
                             const float* __restrict__ src1,   // outT [(b,hw),oc] (MODE 1)
                             const f16*  __restrict__ wt,      // [tap][oc][ci]
                             const float* __restrict__ bias,
                             const float* __restrict__ frames, // residual (MODE 1)
                             float* __restrict__ dst) {
  __shared__ f16 patch[3 * 18 * CI];
  int bid = blockIdx.x;
  int wt4 = bid & 3;
  int h = (bid >> 2) & 63;
  int n = bid >> 8;                 // image 0..13, n = f*2 + b
  int b = n & 1;

  const int tot = 3 * 18 * CI;
  for (int idx = threadIdx.x; idx < tot; idx += blockDim.x) {
    int ci = idx % CI;
    int x = (idx / CI) % 18;
    int dy = idx / (CI * 18);
    int gy = h + dy - 1;
    int gx = wt4 * 16 + x - 1;
    float val = 0.f;
    if (gy >= 0 && gy < HH && gx >= 0 && gx < WW) {
      if (MODE == 0 || ci < 64)
        val = src0[(((size_t)n * 64 + ci) * HH + gy) * WW + gx];
      else
        val = src1[((size_t)b * HW + gy * WW + gx) * 64 + (ci - 64)];
    }
    patch[(dy * 18 + x) * CI + ci] = (f16)val;
  }
  __syncthreads();

  int wave = threadIdx.x >> 5;      // 0..3 -> out-channel tile
  int lane = threadIdx.x & 31;
  int m = lane & 15, half = lane >> 4;
  v8f acc = {};
#pragma unroll
  for (int tap = 0; tap < 9; ++tap) {
    int dy = tap / 3, dx = tap % 3;
    for (int cb = 0; cb < CI; cb += 32) {
      v16h a;
#pragma unroll
      for (int v = 0; v < 8; ++v) {
        int k = half * 16 + 2 * v;
        const f16* p = &patch[(dy * 18 + (m + dx)) * CI + cb + k];
        a[2 * v] = p[0];
        a[2 * v + 1] = p[1];
      }
      v16h bf;
#pragma unroll
      for (int v = 0; v < 8; ++v) {
        int k = cb + half * 16 + 2 * v;
        const f16* p = wt + ((size_t)tap * 64 + wave * 16 + m) * CI + k;
        bf[2 * v] = p[0];
        bf[2 * v + 1] = p[1];
      }
      acc = wmma16(a, bf, acc);
    }
  }
  int oc = wave * 16 + (lane & 15);
  float bv = bias[oc];
#pragma unroll
  for (int v = 0; v < 8; ++v) {
    int px = v + half * 8;
    int gx = wt4 * 16 + px;
    float x = acc[v] + bv;
    x = (x >= 0.f) ? x : NEG_SLOPE * x;
    size_t o = (((size_t)n * 64 + oc) * HH + h) * WW + gx;
    dst[o] = (MODE == 0) ? x : (x + frames[o]);
  }
}

// ---------------------------------------------------------------------------
// Small helper kernels
// ---------------------------------------------------------------------------
__global__ void prep_wt3(const float* __restrict__ w, f16* __restrict__ o) {
  int i = blockIdx.x * blockDim.x + threadIdx.x;    // 9*64*64
  if (i >= 9 * 64 * 64) return;
  int ci = i & 63, oc = (i >> 6) & 63, tap = i >> 12;
  o[((size_t)tap * 64 + oc) * 64 + ci] = (f16)w[((size_t)oc * 64 + ci) * 9 + tap];
}
__global__ void prep_wt13(const float* __restrict__ w, f16* __restrict__ o) {
  int i = blockIdx.x * blockDim.x + threadIdx.x;    // 9*64*128
  if (i >= 9 * 64 * 128) return;
  int ci = i & 127, oc = (i >> 7) & 63, tap = i >> 13;
  o[((size_t)tap * 64 + oc) * 128 + ci] = (f16)w[((size_t)oc * 128 + ci) * 9 + tap];
}
__global__ void prep_weff(const float* __restrict__ dtw, const float* __restrict__ xpw,
                          float* __restrict__ weff) {
  int i = blockIdx.x * blockDim.x + threadIdx.x;    // 64*64
  if (i >= 64 * 64) return;
  int c = i & 63, d = i >> 6;
  float s = 0.f;
#pragma unroll
  for (int r = 0; r < 4; ++r) s += dtw[d * 4 + r] * xpw[r * 64 + c];
  weff[i] = s;
}
// xflat[(b*L + f*HW + hw), c] = r[(f*2+b), c, hw]
__global__ void make_xflat(const float* __restrict__ r, float* __restrict__ xf) {
  size_t i = (size_t)blockIdx.x * blockDim.x + threadIdx.x;  // BL*64
  if (i >= (size_t)BL * 64) return;
  int c = i & 63;
  size_t row = i >> 6;
  int hw = row & 4095;
  int f = (int)((row >> 12) % 7);
  int b = (int)(row / LL);
  xf[i] = r[(((size_t)(f * 2 + b) * 64 + c) * HW) + hw];
}
// xs = silu(causal depthwise conv1d(xi) + b), xi = xz[:, :64]
__global__ void conv1d_silu(const float* __restrict__ xz, const float* __restrict__ w,
                            const float* __restrict__ b, float* __restrict__ xs) {
  size_t i = (size_t)blockIdx.x * blockDim.x + threadIdx.x;  // BL*64
  if (i >= (size_t)BL * 64) return;
  int d = i & 63;
  size_t blr = i >> 6;
  int l = (int)(blr % LL);
  size_t base = (blr - l) * 128;
  float acc = b[d];
  if (l >= 2) acc += w[d * 3 + 0] * xz[base + (size_t)(l - 2) * 128 + d];
  if (l >= 1) acc += w[d * 3 + 1] * xz[base + (size_t)(l - 1) * 128 + d];
  acc += w[d * 3 + 2] * xz[base + (size_t)l * 128 + d];
  xs[i] = acc / (1.f + expf(-acc));
}

// ---------------------------------------------------------------------------
// Chunked selective scan: h_l = dA_l*h_{l-1} + dBx_l is a linear recurrence.
// Pass A: per (b,d,chunk,n) compute P = prod(dA), S = chunk-local state.
// Pass B: sequentially compose 112 chunk summaries -> entry state per chunk.
// Pass C: replay chunks from entry state, emit y = sum_n h*C_n.
// ---------------------------------------------------------------------------
__global__ void scan_passA(const float* __restrict__ dt, const float* __restrict__ bc,
                           const float* __restrict__ xs, const float* __restrict__ A_log,
                           float* __restrict__ cP, float* __restrict__ cS) {
  int tid = blockIdx.x * blockDim.x + threadIdx.x;
  int gid = tid >> 4;                       // (b,d,chunk)
  if (gid >= BB * DM * NC) return;
  int n = tid & 15;
  int chunk = gid % NC;
  int d = (gid / NC) & 63;
  int b = gid / (NC * DM);
  float A = -expf(A_log[d * 16 + n]);
  float P = 1.f, S = 0.f;
  size_t row0 = (size_t)b * LL + (size_t)chunk * CH;
  for (int l = 0; l < CH; ++l) {
    size_t row = row0 + l;
    __builtin_prefetch(dt + (row + 8) * 64 + d, 0, 1);
    __builtin_prefetch(xs + (row + 8) * 64 + d, 0, 1);
    float dtv = dt[row * 64 + d];
    float Bv = bc[row * 32 + n];
    float xv = xs[row * 64 + d];
    float dA = expf(dtv * A);
    P *= dA;
    S = dA * S + dtv * Bv * xv;
  }
  cP[(size_t)gid * 16 + n] = P;
  cS[(size_t)gid * 16 + n] = S;
}

__global__ void scan_passB(const float* __restrict__ cP, const float* __restrict__ cS,
                           float* __restrict__ hs) {
  int tid = blockIdx.x * blockDim.x + threadIdx.x;   // B*DM*16 = 2048
  if (tid >= BB * DM * 16) return;
  int n = tid & 15;
  int bd = tid >> 4;
  float h = 0.f;
  for (int c = 0; c < NC; ++c) {
    size_t idx = ((size_t)bd * NC + c) * 16 + n;
    hs[idx] = h;
    h = cP[idx] * h + cS[idx];
  }
}

__global__ void scan_passC(const float* __restrict__ dt, const float* __restrict__ bc,
                           const float* __restrict__ xs, const float* __restrict__ A_log,
                           const float* __restrict__ hs, float* __restrict__ yscan) {
  int tid = blockIdx.x * blockDim.x + threadIdx.x;
  int gid = tid >> 4;
  if (gid >= BB * DM * NC) return;
  int n = tid & 15;
  int chunk = gid % NC;
  int d = (gid / NC) & 63;
  int b = gid / (NC * DM);
  float A = -expf(A_log[d * 16 + n]);
  float h = hs[(size_t)gid * 16 + n];
  size_t row0 = (size_t)b * LL + (size_t)chunk * CH;
  for (int l = 0; l < CH; ++l) {
    size_t row = row0 + l;
    __builtin_prefetch(dt + (row + 8) * 64 + d, 0, 1);
    __builtin_prefetch(xs + (row + 8) * 64 + d, 0, 1);
    float dtv = dt[row * 64 + d];
    float Bv = bc[row * 32 + n];
    float Cv = bc[row * 32 + 16 + n];
    float xv = xs[row * 64 + d];
    h = expf(dtv * A) * h + dtv * Bv * xv;
    float y = h * Cv;
    y += __shfl_xor(y, 1, 32);
    y += __shfl_xor(y, 2, 32);
    y += __shfl_xor(y, 4, 32);
    y += __shfl_xor(y, 8, 32);
    if (n == 0) yscan[row * 64 + d] = y;   // yscan aliases dt: same elem, after reads
  }
}

// y2 = (yscan + xs*D) * silu(z) ; z = xz[:,64+d] ; y2 aliases xs (same element)
__global__ void combine(const float* __restrict__ yscan, const float* __restrict__ xs,
                        const float* __restrict__ xz, const float* __restrict__ Dp,
                        float* __restrict__ y2) {
  size_t i = (size_t)blockIdx.x * blockDim.x + threadIdx.x;  // BL*64
  if (i >= (size_t)BL * 64) return;
  int d = i & 63;
  size_t row = i >> 6;
  float z = xz[row * 128 + 64 + d];
  float sz = z / (1.f + expf(-z));
  y2[i] = (yscan[i] + xs[i] * Dp[d]) * sz;
}
// act448[(b*HW+hw)*448 + c*7+f] = m[(b*L + f*HW + hw)*64 + c]
__global__ void make_act448(const float* __restrict__ m, float* __restrict__ a) {
  size_t i = (size_t)blockIdx.x * blockDim.x + threadIdx.x;  // 8192*448
  if (i >= (size_t)8192 * 448) return;
  int col = (int)(i % 448);
  size_t row = i / 448;
  int c = col / 7, f = col % 7;
  int b = (int)(row >> 12);
  int hw = (int)(row & 4095);
  a[i] = m[((size_t)b * LL + (size_t)f * HW + hw) * 64 + c];
}

// ---------------------------------------------------------------------------
extern "C" void kernel_launch(void* const* d_in, const int* in_sizes, int n_in,
                              void* d_out, int out_size, void* d_ws, size_t ws_size,
                              hipStream_t stream) {
  (void)in_sizes; (void)n_in; (void)out_size; (void)ws_size;
  const float* lf[7];
  for (int i = 0; i < 7; ++i) lf[i] = (const float*)d_in[i];
  const float* conv3_w   = (const float*)d_in[7];
  const float* conv3_b   = (const float*)d_in[8];
  const float* conv1x1_w = (const float*)d_in[9];
  const float* conv1x1_b = (const float*)d_in[10];
  const float* conv13_w  = (const float*)d_in[11];
  const float* conv13_b  = (const float*)d_in[12];
  const float* in_proj_w = (const float*)d_in[13];
  const float* conv1d_w  = (const float*)d_in[14];
  const float* conv1d_b  = (const float*)d_in[15];
  const float* x_proj_w  = (const float*)d_in[16];
  const float* dt_proj_w = (const float*)d_in[17];
  const float* dt_proj_b = (const float*)d_in[18];
  const float* A_log     = (const float*)d_in[19];
  const float* Dp        = (const float*)d_in[20];
  const float* out_proj_w= (const float*)d_in[21];

  float* ws     = (float*)d_ws;
  float* frames = ws + OFF_FRAMES;
  float* r      = ws + OFF_R;
  float* xflat  = ws + OFF_XFLAT;        // reused as act448 later
  float* xz     = ws + OFF_XZ;           // reused as m later
  float* xs     = ws + OFF_XS;           // reused as y2 later
  float* dt     = ws + OFF_DT;           // reused as yscan later
  float* bc     = ws + OFF_BC;           // reused as outT later
  float* weff   = ws + OFF_WEFF;
  f16*   wt3h   = (f16*)(ws + OFF_WT3);
  f16*   wt13h  = (f16*)(ws + OFF_WT13);
  float* cP     = ws + OFF_CP;
  float* cS     = ws + OFF_CS;
  float* hs     = ws + OFF_HS;
  float* dout   = (float*)d_out;

  // 0) stack frames (7,2,64,64,64)
  for (int f = 0; f < 7; ++f)
    hipMemcpyAsync(frames + (size_t)f * 2 * 64 * HW, lf[f],
                   (size_t)2 * 64 * HW * sizeof(float),
                   hipMemcpyDeviceToDevice, stream);

  // 1) weight prep
  prep_wt3 <<<(9 * 64 * 64 + 255) / 256, 256, 0, stream>>>(conv3_w, wt3h);
  prep_wt13<<<(9 * 64 * 128 + 255) / 256, 256, 0, stream>>>(conv13_w, wt13h);
  prep_weff<<<16, 256, 0, stream>>>(dt_proj_w, x_proj_w, weff);

  // 2) conv3 + lrelu -> r
  conv3x3_wmma<64, 0><<<NIMG * 64 * 4, 128, 0, stream>>>(frames, nullptr, wt3h,
                                                         conv3_b, nullptr, r);

  // 3) x_flat (BL,64)
  make_xflat<<<(BL * 64 + 255) / 256, 256, 0, stream>>>(r, xflat);

  // 4) in_proj: (BL,64)x(128,64)^T -> xz (BL,128)
  gemm_wmma<0, 8><<<BL / 16, 32, 0, stream>>>(xflat, in_proj_w, nullptr,
                                              xz, 64, 64, 64, 128);

  // 5) depthwise causal conv1d + silu -> xs
  conv1d_silu<<<(BL * 64 + 255) / 256, 256, 0, stream>>>(xz, conv1d_w, conv1d_b, xs);

  // 6) dt = softplus(xs @ Weff^T + b)
  gemm_wmma<1, 4><<<BL / 16, 32, 0, stream>>>(xs, weff, dt_proj_b,
                                              dt, 64, 64, 64, 64);

  // 7) B,C = xs @ x_proj_w[4:36]^T -> bc (BL,32)
  gemm_wmma<0, 2><<<BL / 16, 32, 0, stream>>>(xs, x_proj_w + 4 * 64,
                                              nullptr, bc, 64, 64, 64, 32);

  // 8) chunked selective scan -> yscan (aliases dt)
  scan_passA<<<(BB * DM * NC * 16 + 255) / 256, 256, 0, stream>>>(dt, bc, xs, A_log,
                                                                  cP, cS);
  scan_passB<<<(BB * DM * 16 + 255) / 256, 256, 0, stream>>>(cP, cS, hs);
  scan_passC<<<(BB * DM * NC * 16 + 255) / 256, 256, 0, stream>>>(dt, bc, xs, A_log,
                                                                  hs, dt);

  // 9) combine: y2 = (yscan + xs*D)*silu(z)   (y2 aliases xs)
  combine<<<(BL * 64 + 255) / 256, 256, 0, stream>>>(dt, xs, xz, Dp, xs);

  // 10) out_proj: m = y2 @ out_proj_w^T   (m aliases xz base)
  gemm_wmma<0, 4><<<BL / 16, 32, 0, stream>>>(xs, out_proj_w, nullptr,
                                              xz, 64, 64, 64, 64);

  // 11) act448 [(b,hw) x (c*7+f)]  (aliases xflat)
  make_act448<<<((size_t)8192 * 448 + 255) / 256, 256, 0, stream>>>(xz, xflat);

  // 12) conv1x1 + lrelu: outT (8192,64)  (aliases bc)
  gemm_wmma<2, 4><<<8192 / 16, 32, 0, stream>>>(xflat, conv1x1_w, conv1x1_b,
                                                bc, 448, 448, 448, 64);

  // 13) conv13 on [r | out] + lrelu + residual frames -> d_out
  conv3x3_wmma<128, 1><<<NIMG * 64 * 4, 128, 0, stream>>>(r, bc, wt13h, conv13_b,
                                                          frames, dout);
}
